// MoNet_59270548685351
// MI455X (gfx1250) — matrix operand order repaired and checked by reading, not compiled, not run on hardware
//
#include <hip/hip_runtime.h>
#include <hip/hip_bf16.h>

typedef __attribute__((ext_vector_type(16))) __bf16 v16bf;
typedef __attribute__((ext_vector_type(8)))  float  v8f;

#define H    128
#define KKER 3
#define FCOUT 384

__device__ __forceinline__ void atomAddF(float* p, float v) {
    unsafeAtomicAdd(p, v);   // native global_atomic_add_f32 on gfx1250
}

// ---------------------------------------------------------------- utilities
__global__ void k_zero(float* __restrict__ p, int n) {
    int i = blockIdx.x * blockDim.x + threadIdx.x;
    if (i < n) p[i] = 0.0f;
}

__global__ void k_f32_to_bf16(const float* __restrict__ in, __bf16* __restrict__ out, int n) {
    int i = blockIdx.x * blockDim.x + threadIdx.x;
    if (i < n) out[i] = (__bf16)in[i];
}

__global__ void k_deg(const int* __restrict__ dst, float* __restrict__ deg, int E) {
    int i = blockIdx.x * blockDim.x + threadIdx.x;
    if (i < E) atomAddF(&deg[dst[i]], 1.0f);
}

// ---------------------------------------------------------------- weight pack
// Re-layout W[128 x NOUT] (f32) into bf16 WMMA B-fragments:
//   Wp[((ntile*4 + kchunk)*32 + lane)*16 + e]
// so each lane's 16 operand elements are 32 contiguous, 32B-aligned bytes.
// B 32x16 fragment (this code's contiguous k order, same as A): lane = column n0+lane&15;
// lanes 0-15 hold K=k0..k0+15, lanes 16-31 hold K=k0+16..k0+31.
template<int NOUT>
__global__ void k_pack_w(const float* __restrict__ W, __bf16* __restrict__ Wp)
{
    const int ntiles = NOUT / 16;
    int idx = blockIdx.x * blockDim.x + threadIdx.x;   // (nt, kc, lane)
    if (idx >= ntiles * 4 * 32) return;
    int lane = idx & 31;
    int kc   = (idx >> 5) & 3;
    int nt   = idx >> 7;
    int hi = lane >> 4, lanelo = lane & 15;
    int n  = nt * 16 + lanelo;
    int kb = kc * 32 + hi * 16;
    __bf16* o = Wp + (size_t)idx * 16;
#pragma unroll
    for (int e = 0; e < 16; ++e)
        o[e] = (__bf16)W[(size_t)(kb + e) * NOUT + n];
}

// ---------------------------------------------------------------- WMMA GEMM
// C[N x NOUT] = A[N x 128](bf16) @ W(pre-packed bf16) + bias, optional bf16 copy.
// One wave computes a 16 x (16*NTB) strip: A fragment loaded once per k-chunk,
// reused for NTB back-to-back v_wmma_f32_16x16x32_bf16 issues.
template<int NOUT, int NTB>
__global__ void k_gemm_wmma(const __bf16* __restrict__ A,
                            const __bf16* __restrict__ Wp,
                            const float*  __restrict__ bias,
                            float*        __restrict__ C,
                            __bf16*       __restrict__ Cbf,
                            int N)
{
    const int ntiles  = NOUT / 16;
    const int ngroups = ntiles / NTB;
    const int mtiles  = N >> 4;
    int wave = blockIdx.x * (blockDim.x >> 5) + (threadIdx.x >> 5);
    if (wave >= mtiles * ngroups) return;          // wave-uniform exit
    int m0  = (wave / ngroups) << 4;
    int nt0 = (wave % ngroups) * NTB;
    int lane   = threadIdx.x & 31;
    int hi     = lane >> 4;
    int lanelo = lane & 15;

    const __bf16* arow = A + (size_t)(m0 + lanelo) * H;
    v8f acc[NTB] = {};
#pragma unroll
    for (int kc = 0; kc < 4; ++kc) {
        // A 16x32 bf16 fragment: lanes 0-15 K=0..15 ; lanes 16-31 K=16..31
        v16bf a;
        int ka = kc * 32 + hi * 16;
#pragma unroll
        for (int e = 0; e < 8; ++e) a[e]     = arow[ka + e];
#pragma unroll
        for (int e = 0; e < 8; ++e) a[8 + e] = arow[ka + 8 + e];
#pragma unroll
        for (int j = 0; j < NTB; ++j) {
            const v16bf b = *(const v16bf*)(Wp + ((size_t)((nt0 + j) * 4 + kc) * 32 + lane) * 16);
            acc[j] = __builtin_amdgcn_wmma_f32_16x16x32_bf16(false, a, false, b,
                                                             (short)0, acc[j], false, false);
        }
    }
    // C/D layout: VGPR r -> row m0 + 8*hi + r, col n0 + lanelo
    int cm = m0 + hi * 8;
#pragma unroll
    for (int j = 0; j < NTB; ++j) {
        int cn = (nt0 + j) * 16 + lanelo;
        float bv = bias[cn];
#pragma unroll
        for (int r = 0; r < 8; ++r) {
            float v = acc[j][r] + bv;
            size_t off = (size_t)(cm + r) * NOUT + cn;
            C[off] = v;
            if (Cbf) Cbf[off] = (__bf16)v;
        }
    }
}

// ---------------------------------------------------------------- edge stage
// One wave per edge: Gaussian-kernel weights (edge-uniform, in registers),
// gather hp[src] (3 x float4 per lane = 128 channels / wave),
// atomic scatter-add into hnew[dst].
__global__ void k_edge(const float* __restrict__ hp,
                       const int*   __restrict__ src,
                       const int*   __restrict__ dst,
                       const float* __restrict__ deg,
                       const float* __restrict__ ppW,   // [2,2]
                       const float* __restrict__ ppb,   // [2]
                       const float* __restrict__ mu,    // [3,2]
                       const float* __restrict__ isig,  // [3,2]
                       float*       __restrict__ hnew,
                       int E)
{
    int wave = blockIdx.x * (blockDim.x >> 5) + (threadIdx.x >> 5);
    if (wave >= E) return;
    int lane = threadIdx.x & 31;
    int s = src[wave];
    int d = dst[wave];
    float ps0 = rsqrtf(deg[s] + 1.0f);
    float ps1 = rsqrtf(deg[d] + 1.0f);
    float p0 = tanhf(ps0 * ppW[0] + ps1 * ppW[2] + ppb[0]);
    float p1 = tanhf(ps0 * ppW[1] + ps1 * ppW[3] + ppb[1]);
    float g[KKER];
#pragma unroll
    for (int k = 0; k < KKER; ++k) {
        float d0 = (p0 - mu[2 * k])     * isig[2 * k];
        float d1 = (p1 - mu[2 * k + 1]) * isig[2 * k + 1];
        g[k] = __expf(-0.5f * (d0 * d0 + d1 * d1));
    }
    const float4* row = (const float4*)(hp + (size_t)s * FCOUT);
    float4 a0 = row[lane];
    float4 a1 = row[32 + lane];
    float4 a2 = row[64 + lane];
    float4 m;
    m.x = g[0] * a0.x + g[1] * a1.x + g[2] * a2.x;
    m.y = g[0] * a0.y + g[1] * a1.y + g[2] * a2.y;
    m.z = g[0] * a0.z + g[1] * a1.z + g[2] * a2.z;
    m.w = g[0] * a0.w + g[1] * a1.w + g[2] * a2.w;
    float* o = hnew + (size_t)d * H + lane * 4;
    atomAddF(o + 0, m.x);
    atomAddF(o + 1, m.y);
    atomAddF(o + 2, m.z);
    atomAddF(o + 3, m.w);
}

// ---------------------------------------------------------------- batch norm
__global__ void k_bn_stats(const float* __restrict__ h, float* __restrict__ sums, int N)
{
    __shared__ float ls[H], lss[H];
    int t = threadIdx.x;                 // 256 threads: 2 rows x 128 channels
    if (t < H) { ls[t] = 0.0f; lss[t] = 0.0f; }
    __syncthreads();
    int ch = t & (H - 1);
    int r0 = blockIdx.x * 2 + (t >> 7);
    int rs = gridDim.x * 2;
    float s = 0.0f, q = 0.0f;
    for (int r = r0; r < N; r += rs) {
        float v = h[(size_t)r * H + ch];
        s += v; q += v * v;
    }
    atomicAdd(&ls[ch], s);               // ds_add_f32
    atomicAdd(&lss[ch], q);
    __syncthreads();
    if (t < H) { atomAddF(&sums[t], ls[t]); atomAddF(&sums[H + t], lss[t]); }
}

__global__ void k_bn_apply(const float* __restrict__ hnew, const float* __restrict__ sums,
                           const float* __restrict__ gamma, const float* __restrict__ beta,
                           float* __restrict__ h, __bf16* __restrict__ hbf, int N)
{
    int i = blockIdx.x * blockDim.x + threadIdx.x;
    if (i >= N * H) return;
    int ch = i & (H - 1);
    float invN = 1.0f / (float)N;
    float mean = sums[ch] * invN;
    float var  = sums[H + ch] * invN - mean * mean;
    float v = (hnew[i] - mean) * rsqrtf(var + 1e-5f) * gamma[ch] + beta[ch];
    v = fmaxf(v, 0.0f);
    float r = h[i] + v;                  // residual
    h[i]   = r;
    hbf[i] = (__bf16)r;
}

// ---------------------------------------------------------------- readout
__global__ void k_readout(const float* __restrict__ h, const int* __restrict__ gid,
                          float* __restrict__ hg, float* __restrict__ cnts, int N)
{
    int wave = blockIdx.x * (blockDim.x >> 5) + (threadIdx.x >> 5);
    if (wave >= N) return;
    int lane = threadIdx.x & 31;
    int g = gid[wave];
    const float4* row = (const float4*)(h + (size_t)wave * H);
    float4 v = row[lane];
    float* o = hg + (size_t)g * H + lane * 4;
    atomAddF(o + 0, v.x); atomAddF(o + 1, v.y);
    atomAddF(o + 2, v.z); atomAddF(o + 3, v.w);
    if (lane == 0) atomAddF(&cnts[g], 1.0f);
}

__global__ void k_mlp(const float* __restrict__ hg, const float* __restrict__ cnts,
                      const float* __restrict__ W1, const float* __restrict__ b1,
                      const float* __restrict__ W2, const float* __restrict__ b2,
                      const float* __restrict__ W3, const float* __restrict__ b3,
                      float* __restrict__ out)
{
    __shared__ float v0[H], v1[64], v2[32];
    int g = blockIdx.x;
    int t = threadIdx.x;
    float inv = 1.0f / cnts[g];
    if (t < H) v0[t] = hg[(size_t)g * H + t] * inv;
    __syncthreads();
    if (t < 64) {
        float a = b1[t];
        for (int i = 0; i < H; ++i) a += v0[i] * W1[i * 64 + t];
        v1[t] = fmaxf(a, 0.0f);
    }
    __syncthreads();
    if (t < 32) {
        float a = b2[t];
        for (int i = 0; i < 64; ++i) a += v1[i] * W2[i * 32 + t];
        v2[t] = fmaxf(a, 0.0f);
    }
    __syncthreads();
    if (t < 10) {
        float a = b3[t];
        for (int i = 0; i < 32; ++i) a += v2[i] * W3[i * 10 + t];
        out[(size_t)g * 10 + t] = a;
    }
}

// ---------------------------------------------------------------- launch
extern "C" void kernel_launch(void* const* d_in, const int* in_sizes, int n_in,
                              void* d_out, int out_size, void* d_ws, size_t ws_size,
                              hipStream_t stream)
{
    const float* x     = (const float*)d_in[0];
    const float* embW  = (const float*)d_in[1];
    const float* embB  = (const float*)d_in[2];
    const float* fcW   = (const float*)d_in[3];
    const float* fcB   = (const float*)d_in[4];
    const float* mu    = (const float*)d_in[5];
    const float* isig  = (const float*)d_in[6];
    const float* gamma = (const float*)d_in[7];
    const float* beta  = (const float*)d_in[8];
    const float* ppW   = (const float*)d_in[9];
    const float* ppb   = (const float*)d_in[10];
    const float* W1    = (const float*)d_in[11];
    const float* b1    = (const float*)d_in[12];
    const float* W2    = (const float*)d_in[13];
    const float* b2    = (const float*)d_in[14];
    const float* W3    = (const float*)d_in[15];
    const float* b3    = (const float*)d_in[16];
    const int*   src   = (const int*)d_in[17];
    const int*   dst   = (const int*)d_in[18];
    const int*   gid   = (const int*)d_in[19];

    int N = in_sizes[0] / H;       // 50000
    int E = in_sizes[17];          // 800000
    int G = out_size / 10;         // 50
    int L = in_sizes[4] / FCOUT;   // 4
    int NH = N * H;
    (void)n_in; (void)ws_size;

    // workspace layout
    char*  ws  = (char*)d_ws;
    size_t off = 0;
    auto alloc = [&](size_t bytes) -> void* {
        void* p = ws + off;
        off += (bytes + 255) & ~(size_t)255;
        return p;
    };
    float*  deg   = (float*) alloc((size_t)N * 4);
    float*  h     = (float*) alloc((size_t)NH * 4);
    __bf16* hbf   = (__bf16*)alloc((size_t)NH * 2);
    __bf16* xbf   = (__bf16*)alloc((size_t)NH * 2);
    float*  hp    = (float*) alloc((size_t)N * FCOUT * 4);
    float*  hnew  = (float*) alloc((size_t)NH * 4);
    float*  sums  = (float*) alloc(2 * H * 4);
    float*  hg    = (float*) alloc((size_t)G * H * 4);
    float*  cnts  = (float*) alloc((size_t)G * 4);
    __bf16* wpEmb = (__bf16*)alloc((size_t)(H / 16)     * 4 * 32 * 16 * 2);
    __bf16* wpFc  = (__bf16*)alloc((size_t)(FCOUT / 16) * 4 * 32 * 16 * 2);

    // degrees
    k_zero<<<(N + 255) / 256, 256, 0, stream>>>(deg, N);
    k_deg <<<(E + 255) / 256, 256, 0, stream>>>(dst, deg, E);

    // node embedding: h = x @ embW + embB  (bf16 WMMA, f32 accumulate)
    k_f32_to_bf16<<<(NH + 255) / 256, 256, 0, stream>>>(x, xbf, NH);
    k_pack_w<H><<<((H / 16) * 128 + 255) / 256, 256, 0, stream>>>(embW, wpEmb);
    {
        int waves = (N / 16) * ((H / 16) / 4);
        k_gemm_wmma<H, 4><<<(waves + 7) / 8, 256, 0, stream>>>(xbf, wpEmb, embB, h, hbf, N);
    }

    for (int i = 0; i < L; ++i) {
        k_pack_w<FCOUT><<<((FCOUT / 16) * 128 + 255) / 256, 256, 0, stream>>>(
            fcW + (size_t)i * H * FCOUT, wpFc);
        int waves = (N / 16) * ((FCOUT / 16) / 4);
        k_gemm_wmma<FCOUT, 4><<<(waves + 7) / 8, 256, 0, stream>>>(
            hbf, wpFc, fcB + (size_t)i * FCOUT, hp, nullptr, N);
        k_zero<<<(NH + 255) / 256, 256, 0, stream>>>(hnew, NH);
        k_edge<<<(E + 7) / 8, 256, 0, stream>>>(
            hp, src, dst, deg,
            ppW + (size_t)i * 4, ppb + (size_t)i * 2,
            mu + (size_t)i * KKER * 2, isig + (size_t)i * KKER * 2,
            hnew, E);
        k_zero<<<1, 256, 0, stream>>>(sums, 2 * H);
        k_bn_stats<<<400, 256, 0, stream>>>(hnew, sums, N);
        k_bn_apply<<<(NH + 255) / 256, 256, 0, stream>>>(
            hnew, sums, gamma + (size_t)i * H, beta + (size_t)i * H, h, hbf, N);
    }

    // per-graph mean readout + MLP head
    k_zero<<<(G * H + 255) / 256, 256, 0, stream>>>(hg, G * H);
    k_zero<<<1, 64, 0, stream>>>(cnts, G);
    k_readout<<<(N + 7) / 8, 256, 0, stream>>>(h, gid, hg, cnts, N);
    k_mlp<<<G, 128, 0, stream>>>(hg, cnts, W1, b1, W2, b2, W3, b3, (float*)d_out);
}
